// MultiHeadAttentionBlock_45354854646384
// MI455X (gfx1250) — matrix operand order repaired and checked by reading, not compiled, not run on hardware
//
#include <hip/hip_runtime.h>

// MI455X / gfx1250 multi-head attention block:
//  - all matmuls on v_wmma_f32_16x16x32_f16 (f16 inputs, fp32 accum)
//  - flash-attention (online softmax), no SxS intermediate in HBM
//  - V stored transposed so every WMMA B-operand is K-contiguous
//  - K/V tiles double-buffered via global_load_async_to_lds_b128 (ASYNCcnt)

typedef __attribute__((ext_vector_type(16))) _Float16 v16h;
typedef __attribute__((ext_vector_type(8)))  float    v8f;

#define D_MODEL 1024
#define NH      16
#define DKH     64
#define SEQ     2048
#define BATCH   4

#define BM  128
#define BN  128
#define BKK 32
#define LDH 40   // LDS row stride in halves (32 data + 8 pad -> 80B, kills bank conflicts)
#define LDP 72   // attention LDS row stride in halves (64 data + 8 pad -> 144B)

// log2(e) / sqrt(d_k): folds 1/8 score scale AND exp->exp2 into the Q projection
#define QSCALE 0.18033688011112043f

__device__ __forceinline__ v8f zero8() {
    v8f z;
    #pragma unroll
    for (int i = 0; i < 8; i++) z[i] = 0.0f;
    return z;
}

// Load a v16h fragment as two 16-byte chunks (lowers to 2x b128 loads).
__device__ __forceinline__ v16h load_frag(const _Float16* p0, const _Float16* p1) {
    union { v16h h; uint4 q[2]; } u;
    __builtin_memcpy(&u.q[0], p0, 16);
    __builtin_memcpy(&u.q[1], p1, 16);
    return u.h;
}

// Async copy 64B global -> LDS (4x b128). INST_OFFSET applies to both the
// LDS and global address, so one address pair covers all four chunks.
// Completion tracked by ASYNCcnt; loads complete in order.
__device__ __forceinline__ void async_copy64(const _Float16* g, _Float16* l) {
    unsigned la = (unsigned)(unsigned long long)(void*)l;   // flat LDS aperture: low 32 bits = LDS byte addr
    asm volatile(
        "global_load_async_to_lds_b128 %0, %1, off\n\t"
        "global_load_async_to_lds_b128 %0, %1, off offset:16\n\t"
        "global_load_async_to_lds_b128 %0, %1, off offset:32\n\t"
        "global_load_async_to_lds_b128 %0, %1, off offset:48"
        :: "v"(la), "v"(g)
        : "memory");
}

// Async copy 32B global -> LDS (2x b128).
__device__ __forceinline__ void async_copy32(const _Float16* g, _Float16* l) {
    unsigned la = (unsigned)(unsigned long long)(void*)l;
    asm volatile(
        "global_load_async_to_lds_b128 %0, %1, off\n\t"
        "global_load_async_to_lds_b128 %0, %1, off offset:16"
        :: "v"(la), "v"(g)
        : "memory");
}

// Stage 16 fp32 -> 16 f16 halves into LDS (2x 16B LDS stores).
__device__ __forceinline__ void stage16_f32(const float* __restrict__ s, _Float16* d) {
    float4 f0, f1, f2, f3;
    __builtin_memcpy(&f0, s + 0,  16);
    __builtin_memcpy(&f1, s + 4,  16);
    __builtin_memcpy(&f2, s + 8,  16);
    __builtin_memcpy(&f3, s + 12, 16);
    union { _Float16 h[8]; uint4 q; } a, b;
    a.h[0] = (_Float16)f0.x; a.h[1] = (_Float16)f0.y; a.h[2] = (_Float16)f0.z; a.h[3] = (_Float16)f0.w;
    a.h[4] = (_Float16)f1.x; a.h[5] = (_Float16)f1.y; a.h[6] = (_Float16)f1.z; a.h[7] = (_Float16)f1.w;
    b.h[0] = (_Float16)f2.x; b.h[1] = (_Float16)f2.y; b.h[2] = (_Float16)f2.z; b.h[3] = (_Float16)f2.w;
    b.h[4] = (_Float16)f3.x; b.h[5] = (_Float16)f3.y; b.h[6] = (_Float16)f3.z; b.h[7] = (_Float16)f3.w;
    __builtin_memcpy(d,     &a.q, 16);
    __builtin_memcpy(d + 8, &b.q, 16);
}

// ---------------------------------------------------------------------------
// C[M=8192, N=1024] = A[M,1024] * W[N,1024]^T + bias
// mode 0: A=f32 query -> q  f16 [b,h,s,d] * QSCALE
// mode 1: A=f32 key   -> k  f16 [b,h,s,d]
// mode 2: A=f32 value -> vT f16 [b,h,d,s]   (transposed for PV GEMM)
// mode 3: A=f16 ctx   -> out f32 [b,s,1024]
// ---------------------------------------------------------------------------
__global__ __launch_bounds__(256)
void proj_gemm(const float* __restrict__ Af,
               const _Float16* __restrict__ Ah,
               const float* __restrict__ Wt,
               const float* __restrict__ bias,
               _Float16* __restrict__ outH,
               float* __restrict__ outF,
               int mode)
{
    __shared__ _Float16 aT[BM * LDH];
    __shared__ _Float16 bT[BN * LDH];

    const int m0   = blockIdx.x * BM;
    const int n0   = blockIdx.y * BN;
    const int tid  = threadIdx.x;
    const int lane = tid & 31;
    const int wave = tid >> 5;
    const int wm   = wave & 3;          // 4 waves along M (32 rows each)
    const int wn   = wave >> 2;         // 2 waves along N (64 cols each)
    const int l15  = lane & 15;
    const int lh8  = (lane >> 4) * 8;   // A-fragment half offset
    const int lh16 = (lane >> 4) * 16;  // unused; B fragments also use lh8

    v8f acc[2][4];
    #pragma unroll
    for (int i = 0; i < 2; i++)
        #pragma unroll
        for (int j = 0; j < 4; j++) acc[i][j] = zero8();

    const int srow = tid >> 1;
    const int sseg = tid & 1;

    for (int k0 = 0; k0 < D_MODEL; k0 += BKK) {
        __syncthreads();
        // stage A tile (128 x 32 halves)
        if (mode < 3) {
            const float* s = Af + (size_t)(m0 + srow) * D_MODEL + k0 + sseg * 16;
            stage16_f32(s, &aT[srow * LDH + sseg * 16]);
            if (k0 + BKK < D_MODEL) __builtin_prefetch(s + BKK, 0, 3);
        } else {
            const _Float16* s = Ah + (size_t)(m0 + srow) * D_MODEL + k0 + sseg * 16;
            async_copy32(s, &aT[srow * LDH + sseg * 16]);
            if (k0 + BKK < D_MODEL) __builtin_prefetch(s + BKK, 0, 3);
        }
        // stage W tile (128 x 32 halves); W rows are K-contiguous (x @ W^T)
        {
            const float* s = Wt + (size_t)(n0 + srow) * D_MODEL + k0 + sseg * 16;
            stage16_f32(s, &bT[srow * LDH + sseg * 16]);
            if (k0 + BKK < D_MODEL) __builtin_prefetch(s + BKK, 0, 3);
        }
        if (mode == 3) asm volatile("s_wait_asynccnt 0x0" ::: "memory");
        __syncthreads();

        v16h afr[2];
        #pragma unroll
        for (int tm = 0; tm < 2; tm++) {
            const _Float16* base = &aT[(wm * 32 + tm * 16 + l15) * LDH + lh8];
            afr[tm] = load_frag(base, base + 16);   // K runs {0..7,16..23} / {8..15,24..31}
        }
        v16h bfr[4];
        #pragma unroll
        for (int tn = 0; tn < 4; tn++) {
            const _Float16* base = &bT[(wn * 64 + tn * 16 + l15) * LDH + lh8];
            bfr[tn] = load_frag(base, base + 16);   // K runs {0..7,16..23} / {8..15,24..31}
        }
        #pragma unroll
        for (int tm = 0; tm < 2; tm++)
            #pragma unroll
            for (int tn = 0; tn < 4; tn++)
                acc[tm][tn] = __builtin_amdgcn_wmma_f32_16x16x32_f16(
                    false, afr[tm], false, bfr[tn], (short)0, acc[tm][tn], false, false);
    }

    // epilogue: C-layout -> lane owns col n (l15), rows r + 8*(lane>=16)
    #pragma unroll
    for (int tm = 0; tm < 2; tm++) {
        #pragma unroll
        for (int tn = 0; tn < 4; tn++) {
            const int   ncol  = n0 + wn * 64 + tn * 16 + l15;
            const float bvv   = bias[ncol];
            const int   mbase = m0 + wm * 32 + tm * 16 + lh8;
            #pragma unroll
            for (int r = 0; r < 8; r++) {
                const int   m  = mbase + r;
                const float v  = acc[tm][tn][r] + bvv;
                const int   bb = m >> 11;
                const int   ss = m & (SEQ - 1);
                if (mode == 0) {
                    outH[((((size_t)bb * NH + (ncol >> 6)) * SEQ + ss) << 6) + (ncol & 63)] =
                        (_Float16)(v * QSCALE);
                } else if (mode == 1) {
                    outH[((((size_t)bb * NH + (ncol >> 6)) * SEQ + ss) << 6) + (ncol & 63)] =
                        (_Float16)v;
                } else if (mode == 2) {
                    outH[(((size_t)bb * NH + (ncol >> 6)) * DKH + (ncol & 63)) * SEQ + ss] =
                        (_Float16)v;
                } else {
                    outF[(size_t)m * D_MODEL + ncol] = v;
                }
            }
        }
    }
}

// ---------------------------------------------------------------------------
// Flash attention: grid (SEQ/64, B*H), block 128 (4 waves x 16 query rows).
// Q is pre-scaled by log2e/8, so softmax uses exp2.
// K/V tiles double-buffered with async global->LDS copies.
// ---------------------------------------------------------------------------
__global__ __launch_bounds__(128)
void flash_attn(const _Float16* __restrict__ qf,
                const _Float16* __restrict__ kf,
                const _Float16* __restrict__ vT,
                _Float16* __restrict__ ctx)
{
    __shared__ _Float16 kTb[2][64 * LDP];    // 64 keys x 64 d, double buffered
    __shared__ _Float16 vTb[2][64 * LDP];    // 64 d x 64 t, double buffered
    __shared__ _Float16 pT[64 * LDP];        // 4 waves x 16 q-rows x 64 t

    const int bh   = blockIdx.y;             // b*NH + h
    const int q0   = blockIdx.x * 64;
    const int tid  = threadIdx.x;
    const int lane = tid & 31;
    const int wave = tid >> 5;
    const int l15  = lane & 15;
    const int lh8  = (lane >> 4) * 8;
    const int lh16 = (lane >> 4) * 16;

    const _Float16* qbase = qf + (size_t)bh * SEQ * DKH;
    const _Float16* kbase = kf + (size_t)bh * SEQ * DKH;
    const _Float16* vbase = vT + (size_t)bh * DKH * SEQ;

    // staging map: 64 rows x 2 segments of 64B; each thread async-copies 64B of K and 64B of V
    const int srow = tid >> 1;
    const int sseg = tid & 1;

    // prologue: start fetching tile 0 into buffer 0 (8 async b128 per thread)
    async_copy64(kbase + (size_t)srow * DKH + sseg * 32, &kTb[0][srow * LDP + sseg * 32]);
    async_copy64(vbase + (size_t)srow * SEQ + sseg * 32, &vTb[0][srow * LDP + sseg * 32]);

    // Q fragments for this wave's 16 rows, both d k-steps, loaded once.
    v16h qfr[2];
    {
        const _Float16* qr = qbase + (size_t)(q0 + wave * 16 + l15) * DKH;
        #pragma unroll
        for (int ks = 0; ks < 2; ks++) {
            const _Float16* p = qr + ks * 32 + lh8;
            qfr[ks] = load_frag(p, p + 16);
        }
    }

    v8f acc[4];
    #pragma unroll
    for (int dt = 0; dt < 4; dt++) acc[dt] = zero8();
    float mrow[8], lrow[8];
    #pragma unroll
    for (int r = 0; r < 8; r++) { mrow[r] = -3.0e38f; lrow[r] = 0.0f; }

    for (int jb = 0; jb < SEQ / 64; jb++) {
        const int cur = jb & 1;
        // all waves done computing from the other buffer -> safe to refill it
        __syncthreads();
        if (jb + 1 < SEQ / 64) {
            const int jn = (jb + 1) * 64;
            async_copy64(kbase + (size_t)(jn + srow) * DKH + sseg * 32,
                         &kTb[1 - cur][srow * LDP + sseg * 32]);
            async_copy64(vbase + (size_t)srow * SEQ + jn + sseg * 32,
                         &vTb[1 - cur][srow * LDP + sseg * 32]);
            // 16 outstanding; async loads complete in order -> <=8 means tile `cur` landed
            asm volatile("s_wait_asynccnt 0x8" ::: "memory");
        } else {
            asm volatile("s_wait_asynccnt 0x0" ::: "memory");
        }
        __syncthreads();   // everyone's chunks of tile `cur` visible

        const _Float16* kc = kTb[cur];
        const _Float16* vc = vTb[cur];

        // S = Q * K^T  (4 key sub-tiles x 2 d k-steps)
        v8f sa[4];
        #pragma unroll
        for (int jt = 0; jt < 4; jt++) {
            sa[jt] = zero8();
            #pragma unroll
            for (int ks = 0; ks < 2; ks++) {
                const _Float16* kb = &kc[(jt * 16 + l15) * LDP + ks * 32 + lh8];
                v16h bf = load_frag(kb, kb + 16);
                sa[jt] = __builtin_amdgcn_wmma_f32_16x16x32_f16(
                    false, qfr[ks], false, bf, (short)0, sa[jt], false, false);
            }
        }

        // online softmax (base-2). Row stats reduced across the 16-lane N stripe.
        float mnew[8], scl[8];
        #pragma unroll
        for (int r = 0; r < 8; r++) {
            float mv = fmaxf(fmaxf(sa[0][r], sa[1][r]), fmaxf(sa[2][r], sa[3][r]));
            mv = fmaxf(mv, __shfl_xor(mv, 1, 32));
            mv = fmaxf(mv, __shfl_xor(mv, 2, 32));
            mv = fmaxf(mv, __shfl_xor(mv, 4, 32));
            mv = fmaxf(mv, __shfl_xor(mv, 8, 32));
            const float mn = fmaxf(mrow[r], mv);
            mnew[r] = mn;
            scl[r]  = exp2f(mrow[r] - mn);
            mrow[r] = mn;
        }
        #pragma unroll
        for (int dt = 0; dt < 4; dt++)
            #pragma unroll
            for (int r = 0; r < 8; r++) acc[dt][r] *= scl[r];

        #pragma unroll
        for (int r = 0; r < 8; r++) {
            float ps = 0.0f;
            const int prow = wave * 16 + r + lh8;   // C-layout row for this lane
            #pragma unroll
            for (int jt = 0; jt < 4; jt++) {
                const float p = exp2f(sa[jt][r] - mnew[r]);
                ps += p;
                pT[prow * LDP + jt * 16 + l15] = (_Float16)p;
            }
            ps += __shfl_xor(ps, 1, 32);
            ps += __shfl_xor(ps, 2, 32);
            ps += __shfl_xor(ps, 4, 32);
            ps += __shfl_xor(ps, 8, 32);
            lrow[r] = lrow[r] * scl[r] + ps;
        }

        // fence P stores before same-wave A-fragment reloads (per-wave region,
        // no cross-wave barrier needed; DS ops are in-order per wave)
        asm volatile("s_wait_dscnt 0" ::: "memory");

        // O += P * V
        const _Float16* pb = &pT[(wave * 16 + l15) * LDP];
        #pragma unroll
        for (int ks = 0; ks < 2; ks++) {
            const _Float16* pp = pb + ks * 32 + lh8;
            v16h pa = load_frag(pp, pp + 16);
            #pragma unroll
            for (int dt = 0; dt < 4; dt++) {
                const _Float16* vb = &vc[(dt * 16 + l15) * LDP + ks * 32 + lh8];
                v16h bf = load_frag(vb, vb + 16);
                acc[dt] = __builtin_amdgcn_wmma_f32_16x16x32_f16(
                    false, pa, false, bf, (short)0, acc[dt], false, false);
            }
        }
    }

    // normalize and store ctx f16 [b, s, h*64+d]
    const int bb = bh >> 4;
    const int hh = bh & 15;
    #pragma unroll
    for (int dt = 0; dt < 4; dt++) {
        const int col = hh * 64 + dt * 16 + l15;
        #pragma unroll
        for (int r = 0; r < 8; r++) {
            const int   ss = q0 + wave * 16 + r + lh8;
            const float o  = acc[dt][r] / lrow[r];
            ctx[((size_t)bb * SEQ + ss) * D_MODEL + col] = (_Float16)o;
        }
    }
}

// ---------------------------------------------------------------------------
extern "C" void kernel_launch(void* const* d_in, const int* in_sizes, int n_in,
                              void* d_out, int out_size, void* d_ws, size_t ws_size,
                              hipStream_t stream) {
    const float* query = (const float*)d_in[0];
    const float* key_  = (const float*)d_in[1];
    const float* value = (const float*)d_in[2];
    // d_in[3] = mask: intentionally unused (reference discards masked_fill)
    const float* Wq = (const float*)d_in[4];
    const float* bq = (const float*)d_in[5];
    const float* Wk = (const float*)d_in[6];
    const float* bk = (const float*)d_in[7];
    const float* Wv = (const float*)d_in[8];
    const float* bv = (const float*)d_in[9];
    const float* Wo = (const float*)d_in[10];
    const float* bo = (const float*)d_in[11];
    float* out = (float*)d_out;

    _Float16* ws = (_Float16*)d_ws;
    const size_t NE = (size_t)BATCH * NH * SEQ * DKH;   // 8388608 halves (16 MiB)
    _Float16* qfh  = ws;
    _Float16* kfh  = ws + NE;
    _Float16* vTh  = ws + 2 * NE;
    _Float16* ctxh = ws + 3 * NE;                       // total 64 MiB of ws

    const dim3 gridP(BATCH * SEQ / BM, D_MODEL / BN);   // 64 x 8
    proj_gemm<<<gridP, 256, 0, stream>>>(query, nullptr, Wq, bq, qfh, nullptr, 0);
    proj_gemm<<<gridP, 256, 0, stream>>>(key_,  nullptr, Wk, bk, kfh, nullptr, 1);
    proj_gemm<<<gridP, 256, 0, stream>>>(value, nullptr, Wv, bv, vTh, nullptr, 2);

    flash_attn<<<dim3(SEQ / 64, BATCH * NH), 128, 0, stream>>>(qfh, kfh, vTh, ctxh);

    proj_gemm<<<gridP, 256, 0, stream>>>(nullptr, ctxh, Wo, bo, nullptr, out, 3);
}
